// HadamardConv2d_31327491457245
// MI455X (gfx1250) — matrix operand order repaired and checked by reading, not compiled, not run on hardware
//
#include <hip/hip_runtime.h>
#include <stdint.h>

// ---------------------------------------------------------------------------
// CDNA5 (gfx1250) wave32 WMMA bf16 path.  D = A(16x32) * B(32x16) + C(16x16 f32)
// Operand layouts per CDNA5 ISA 7.12.2 (wave32):
//   A: lane(m=L&15,h=L>>4): a.s[j]=A[m][8h+j], a.s[8+j]=A[m][16+8h+j]  (2 runs of 8)
//   B: lane(n=L&15,h=L>>4): b.s[j]=B[8h+j][n], b.s[8+j]=B[16+8h+j][n]
//      -> from TRANSPOSED storage Bt (N-major): b.s[j]=Bt[n][8h+j], b.s[8+j]=Bt[n][16+8h+j]
//   C/D: lane(n=L&15): c[v] = row (v + 8*(L>>4)), col n
// All B-side matrices stored N-major (2x ds_load_b128 per operand); WMMA
// results destined to be B operands stored transposed => packed ds_store_b128.
// Hadamard is symmetric so one LDS copy serves both A and Bt roles.
// bf16 packing via inline-asm v_cvt_pk_bf16_f32 (1 VALU per pair; the clang
// builtin is not declared for gfx1250 on this toolchain).
// Neighbor-lane max via DPP quad_perm [1,0,3,2] (0xB1), no LDS traffic.
// ---------------------------------------------------------------------------

typedef __attribute__((ext_vector_type(16))) __bf16 v16bf;
typedef __attribute__((ext_vector_type(8)))  float  v8f;
typedef unsigned short u16;

union BfVec { v16bf v; u16 s[16]; unsigned u[8]; };

__device__ __forceinline__ unsigned pk(float a, float b) {   // low=cvt(a), high=cvt(b)
  unsigned r;
  asm("v_cvt_pk_bf16_f32 %0, %1, %2" : "=v"(r) : "v"(a), "v"(b));
  return r;
}
__device__ __forceinline__ u16 f2bf(float f) { return (u16)pk(f, 0.f); }

// max with xor-1 neighbor lane: DPP quad_perm:[1,0,3,2]
__device__ __forceinline__ float max_nb(float v) {
  int o = __builtin_amdgcn_mov_dpp(__float_as_int(v), 0xB1, 0xF, 0xF, true);
  return fmaxf(v, __int_as_float(o));
}

__device__ __forceinline__ v8f wmma_bf16(v16bf a, v16bf b, v8f c) {
  return __builtin_amdgcn_wmma_f32_16x16x32_bf16(false, a, false, b, (short)0, c,
                                                 false, false);
}

// A operand from row-major u16 buffer
__device__ __forceinline__ v16bf load_A(const u16* p, int row0, int col0, int stride) {
  const int lane = threadIdx.x & 31;
  const int m = lane & 15, hf = lane >> 4;
  const u16* q0 = p + (row0 + m) * stride + col0 + hf * 8;
  BfVec a;
#pragma unroll
  for (int j = 0; j < 8; ++j) { a.s[j] = q0[j]; a.s[8 + j] = q0[16 + j]; }
  return a.v;
}

// B operand from transposed (N-major) u16 buffer: row n holds K-contiguous data
__device__ __forceinline__ v16bf load_Bt(const u16* p, int nrow0, int kcol0, int stride) {
  const int lane = threadIdx.x & 31;
  const int n = lane & 15, koff = (lane >> 4) << 3;
  const u16* q = p + (nrow0 + n) * stride + kcol0 + koff;
  BfVec b;
#pragma unroll
  for (int j = 0; j < 8; ++j) { b.s[j] = q[j]; b.s[8 + j] = q[16 + j]; }
  return b.v;
}

// pack 8 f32 WMMA results -> uint4 of bf16 pairs (for contiguous b128 store)
__device__ __forceinline__ uint4 pack8(const v8f& c) {
  uint4 r;
  r.x = pk(c[0], c[1]); r.y = pk(c[2], c[3]);
  r.z = pk(c[4], c[5]); r.w = pk(c[6], c[7]);
  return r;
}

// ---------------------------------------------------------------------------
// Prep: spectral kernels K1c=(H^ k H^)/32768 (32x32x32), K2c=/4096 (64x16x16),
// H^[i][j]=(-1)^popc(i&j) (unnormalized Sylvester Hadamard; exact in bf16).
// ---------------------------------------------------------------------------
__global__ __launch_bounds__(256) void prep_filters_kernel(
    const float* __restrict__ f1, const float* __restrict__ f2,
    float* __restrict__ K1c, float* __restrict__ K2c) {
  const int idx = blockIdx.x * 256 + threadIdx.x;
  if (idx < 32 * 1024) {
    const int f = idx >> 10, r = (idx >> 5) & 31, c = idx & 31;
    const float* k = f1 + f * 9;
    float s = 0.f;
#pragma unroll
    for (int a = 0; a < 3; ++a)
#pragma unroll
      for (int b = 0; b < 3; ++b) {
        float sa = (__popc(r & a) & 1) ? -1.f : 1.f;
        float sb = (__popc(b & c) & 1) ? -1.f : 1.f;
        s += sa * k[a * 3 + b] * sb;
      }
    K1c[idx] = s * (1.f / 32768.f);
  }
  const int idx2 = idx - 32 * 1024;
  if (idx2 >= 0 && idx2 < 64 * 256) {
    const int f = idx2 >> 8, r = (idx2 >> 4) & 15, c = idx2 & 15;
    const float* k = f2 + f * 9;
    float s = 0.f;
#pragma unroll
    for (int a = 0; a < 3; ++a)
#pragma unroll
      for (int b = 0; b < 3; ++b) {
        float sa = (__popc(r & a) & 1) ? -1.f : 1.f;
        float sb = (__popc(b & c) & 1) ? -1.f : 1.f;
        s += sa * k[a * 3 + b] * sb;
      }
    K2c[idx2] = s * (1.f / 4096.f);
  }
}

// fc1_w (512,4096) f32 -> bf16, SAME layout (native layout is already B-friendly
// N-major for out = act * fc1_w^T). 8 elements per thread, vectorized.
__global__ __launch_bounds__(256) void prep_weights_kernel(
    const float* __restrict__ fc1_w, u16* __restrict__ Wb) {
  const int i8 = (blockIdx.x * 256 + threadIdx.x) * 8;
  if (i8 < 512 * 4096) {
    float4 a = *(const float4*)(fc1_w + i8);
    float4 b = *(const float4*)(fc1_w + i8 + 4);
    uint4 o;
    o.x = pk(a.x, a.y); o.y = pk(a.z, a.w);
    o.z = pk(b.x, b.y); o.w = pk(b.z, b.w);
    *(uint4*)(Wb + i8) = o;
  }
}

// ---------------------------------------------------------------------------
// Stage 1 (one block per image, 128 thr = 4 waves):
//   T = H^(Sum_c x_c)H^  (cooperative, 2 barriers)
//   Then each wave owns 8 filters end-to-end, ZERO barriers in the loop:
//     O_f = H^(T.*K1c_f)H^ + b1_f ; psum += maxpool2(relu(O_f))
//   Pool: in-lane row-pair max + DPP quad-perm column max.
// ---------------------------------------------------------------------------
__global__ __launch_bounds__(128) void stage1_kernel(
    const float* __restrict__ x, const float* __restrict__ K1c,
    const float* __restrict__ bias1, float* __restrict__ psum1) {
  __shared__ __align__(16) u16   Hb[32 * 32];     // Hadamard (symmetric: A & Bt)
  __shared__ __align__(16) u16   Xb[32 * 32];     // channel-sum, row-major (A)
  __shared__ __align__(16) u16   Ut[32 * 32];     // U^T (Bt for T-phase)
  __shared__ __align__(16) float Tf[32 * 32];     // spectral image, f32
  __shared__ __align__(16) u16   Vt[4][32 * 32];  // per-wave V^T (Bt for O-phase)
  __shared__ __align__(16) float Ps[4][16 * 16];  // per-wave pooled accumulators

  const int b = blockIdx.x, t = threadIdx.x;
  const int wave = t >> 5, lane = t & 31;
  const int m = lane & 15, hf = lane >> 4;
  const float* xb = x + (size_t)b * 3072;

  {  // fill Hb (packed sign writes) and Xb (channel sum), zero Ps
    const int base = t * 8;             // 8 consecutive elements per thread
    const int r = t >> 2, c0 = (t & 3) * 8;
    uint4 hq;
    unsigned hw[4];
#pragma unroll
    for (int p = 0; p < 4; ++p) {
      u16 s0 = (__popc(r & (c0 + 2 * p)) & 1) ? 0xBF80u : 0x3F80u;
      u16 s1 = (__popc(r & (c0 + 2 * p + 1)) & 1) ? 0xBF80u : 0x3F80u;
      hw[p] = ((unsigned)s1 << 16) | s0;
    }
    hq.x = hw[0]; hq.y = hw[1]; hq.z = hw[2]; hq.w = hw[3];
    *(uint4*)(Hb + base) = hq;

    float4 a0 = *(const float4*)(xb + base);
    float4 a1 = *(const float4*)(xb + base + 4);
    float4 b0 = *(const float4*)(xb + 1024 + base);
    float4 b1 = *(const float4*)(xb + 1024 + base + 4);
    float4 c0v = *(const float4*)(xb + 2048 + base);
    float4 c1v = *(const float4*)(xb + 2048 + base + 4);
    uint4 xq;
    xq.x = pk(a0.x + b0.x + c0v.x, a0.y + b0.y + c0v.y);
    xq.y = pk(a0.z + b0.z + c0v.z, a0.w + b0.w + c0v.w);
    xq.z = pk(a1.x + b1.x + c1v.x, a1.y + b1.y + c1v.y);
    xq.w = pk(a1.z + b1.z + c1v.z, a1.w + b1.w + c1v.w);
    *(uint4*)(Xb + base) = xq;

    float* psf = &Ps[0][0];
    for (int i = t; i < 1024; i += 128) psf[i] = 0.f;
  }
  __syncthreads();

  const int tmw = (wave >> 1) << 4, tnw = (wave & 1) << 4;  // cooperative tile
  const int cn = tnw + m, crow0 = tmw + (hf << 3);

  {  // U = X * H ; store U^T (contiguous packed per lane)
    v8f c = {};
    c = wmma_bf16(load_A(Xb, tmw, 0, 32), load_Bt(Hb, tnw, 0, 32), c);
    *(uint4*)(Ut + cn * 32 + crow0) = pack8(c);
  }
  __syncthreads();
  {  // T = H * U ; store f32 row-major
    v8f c = {};
    c = wmma_bf16(load_A(Hb, tmw, 0, 32), load_Bt(Ut, tnw, 0, 32), c);
#pragma unroll
    for (int v = 0; v < 8; ++v) Tf[(crow0 + v) * 32 + cn] = c[v];
  }
  __syncthreads();

  // hoisted Hadamard operands (loop-invariant)
  const v16bf aH0 = load_A(Hb, 0, 0, 32), aH1 = load_A(Hb, 16, 0, 32);
  const v16bf bH0 = load_Bt(Hb, 0, 0, 32), bH1 = load_Bt(Hb, 16, 0, 32);

  // cache this lane's T slice in registers (loop-invariant, 32 floats)
  float4 tcr[2][4];
#pragma unroll
  for (int tmi = 0; tmi < 2; ++tmi) {
    const float* tp = Tf + (tmi * 16 + m) * 32 + hf * 8;
    tcr[tmi][0] = *(const float4*)tp;
    tcr[tmi][1] = *(const float4*)(tp + 4);
    tcr[tmi][2] = *(const float4*)(tp + 16);
    tcr[tmi][3] = *(const float4*)(tp + 20);
  }

  u16* vt = Vt[wave];
  float* ps = Ps[wave];

  for (int fi = 0; fi < 8; ++fi) {
    const int f = wave * 8 + fi;
    const float* Kf = K1c + f * 1024;

    // Build A operands of P = T .* Kf for row blocks 0 and 16, directly per lane
    v16bf aP[2];
#pragma unroll
    for (int tmi = 0; tmi < 2; ++tmi) {
      const float4* kq = (const float4*)(Kf + (tmi * 16 + m) * 32 + hf * 8);
      float4 k0 = kq[0], k1 = kq[1], k2 = kq[4], k3 = kq[5];
      float4 t0 = tcr[tmi][0], t1 = tcr[tmi][1], t2 = tcr[tmi][2], t3 = tcr[tmi][3];
      BfVec a;
      a.u[0] = pk(t0.x * k0.x, t0.y * k0.y);
      a.u[1] = pk(t0.z * k0.z, t0.w * k0.w);
      a.u[2] = pk(t1.x * k1.x, t1.y * k1.y);
      a.u[3] = pk(t1.z * k1.z, t1.w * k1.w);
      a.u[4] = pk(t2.x * k2.x, t2.y * k2.y);
      a.u[5] = pk(t2.z * k2.z, t2.w * k2.w);
      a.u[6] = pk(t3.x * k3.x, t3.y * k3.y);
      a.u[7] = pk(t3.z * k3.z, t3.w * k3.w);
      aP[tmi] = a.v;
    }

    // V = P * H (4 tiles by this wave), store V^T into own scratch
    v8f cv0 = {}, cv1 = {}, cv2 = {}, cv3 = {};
    cv0 = wmma_bf16(aP[0], bH0, cv0);
    cv1 = wmma_bf16(aP[0], bH1, cv1);
    cv2 = wmma_bf16(aP[1], bH0, cv2);
    cv3 = wmma_bf16(aP[1], bH1, cv3);
    *(uint4*)(vt + (0 + m) * 32 + 0 + (hf << 3))  = pack8(cv0);
    *(uint4*)(vt + (16 + m) * 32 + 0 + (hf << 3)) = pack8(cv1);
    *(uint4*)(vt + (0 + m) * 32 + 16 + (hf << 3)) = pack8(cv2);
    *(uint4*)(vt + (16 + m) * 32 + 16 + (hf << 3)) = pack8(cv3);

    // O = H * V (4 tiles), intra-wave LDS RAW (in-order per ISA)
    const v16bf bV0 = load_Bt(vt, 0, 0, 32), bV1 = load_Bt(vt, 16, 0, 32);
    v8f z = {};
    v8f co[4];
    co[0] = wmma_bf16(aH0, bV0, z);
    co[1] = wmma_bf16(aH0, bV1, z);
    co[2] = wmma_bf16(aH1, bV0, z);
    co[3] = wmma_bf16(aH1, bV1, z);

    // bias + relu + 2x2 maxpool (row pairs in-lane, col pairs via DPP)
    const float bf = bias1[f];
#pragma unroll
    for (int q = 0; q < 4; ++q) {
      const int tmq = (q >> 1) << 4, tnq = (q & 1) << 4;
      const int pj = (tnq >> 1) + (m >> 1);
#pragma unroll
      for (int i = 0; i < 4; ++i) {
        float r0v = fmaxf(co[q][2 * i] + bf, 0.f);
        float r1v = fmaxf(co[q][2 * i + 1] + bf, 0.f);
        float mv = max_nb(fmaxf(r0v, r1v));
        if ((m & 1) == 0) {
          const int pi = (tmq >> 1) + (hf << 2) + i;
          ps[pi * 16 + pj] += mv;
        }
      }
    }
  }
  __syncthreads();
  float* outp = psum1 + (size_t)b * 256;
  for (int i = t; i < 256; i += 128) outp[i] = Ps[0][i] + Ps[1][i] + Ps[2][i] + Ps[3][i];
}

// ---------------------------------------------------------------------------
// Stage 2 (one block per image, 4 waves): T2 = H2^ psum H2^ (2 barriers), then
// each wave owns 16 filters, zero barriers: O = H2^(T2.*K2c_f)H2^ + b2 -> relu
// -> pool -> act bf16 (B,4096).  16x16 matmuls as zero-padded 16x16x32 WMMAs.
// ---------------------------------------------------------------------------
__global__ __launch_bounds__(128) void stage2_kernel(
    const float* __restrict__ psum1, const float* __restrict__ K2c,
    const float* __restrict__ bias2, u16* __restrict__ act) {
  __shared__ __align__(16) u16   HA[16 * 32];      // H2^ padded right (A and Bt)
  __shared__ __align__(16) u16   SAr[16 * 32];     // psum as A, padded right
  __shared__ __align__(16) u16   Ut2[16 * 32];     // U2^T padded right (Bt)
  __shared__ __align__(16) float Tf[16 * 16];
  __shared__ __align__(16) u16   VT2[4][16 * 32];  // per-wave V^T padded right

  const int b = blockIdx.x, t = threadIdx.x;
  const int wave = t >> 5, lane = t & 31;
  const int m = lane & 15, hf = lane >> 4;
  const float* S = psum1 + (size_t)b * 256;
  const u16 ONE = 0x3F80u, NEG1 = 0xBF80u, Z = 0;

  for (int idx = t; idx < 512; idx += 128) {
    const int r = idx >> 5, c = idx & 31;
    HA[idx]  = (c < 16) ? ((__popc(r & c) & 1) ? NEG1 : ONE) : Z;
    SAr[idx] = (c < 16) ? f2bf(S[r * 16 + c]) : Z;
    Ut2[idx] = Z;
  }
  for (int idx = t; idx < 4 * 512; idx += 128) {   // static right-half pads
    if (((idx & 511) & 31) >= 16) VT2[idx >> 9][idx & 511] = Z;
  }
  __syncthreads();

  {  // U2 = S * H2 (computed by all waves; wave0 publishes U2^T)
    v8f c = {};
    c = wmma_bf16(load_A(SAr, 0, 0, 32), load_Bt(HA, 0, 0, 32), c);
    if (wave == 0) *(uint4*)(Ut2 + m * 32 + (hf << 3)) = pack8(c);
  }
  __syncthreads();
  {  // T2 = H2 * U2 (wave0 publishes f32)
    v8f c = {};
    c = wmma_bf16(load_A(HA, 0, 0, 32), load_Bt(Ut2, 0, 0, 32), c);
    if (wave == 0) {
#pragma unroll
      for (int v = 0; v < 8; ++v) Tf[((hf << 3) + v) * 16 + m] = c[v];
    }
  }
  __syncthreads();

  const v16bf aHA = load_A(HA, 0, 0, 32);
  const v16bf bHA = load_Bt(HA, 0, 0, 32);
  u16* vt = VT2[wave];

  // cache this lane's T2 slice (loop-invariant, 8 floats)
  const float* tp = Tf + m * 16 + hf * 8;
  const float4 t0 = *(const float4*)tp;
  const float4 t1 = *(const float4*)(tp + 4);

  for (int fi = 0; fi < 16; ++fi) {
    const int f = wave * 16 + fi;
    const float* Kf = K2c + f * 256;

    // A operand of P = T2 .* Kf (cols >=16 are the zero pad)
    const float4* kq = (const float4*)(Kf + m * 16 + hf * 8);
    float4 k0 = kq[0], k1 = kq[1];
    BfVec a;
    a.u[0] = pk(t0.x * k0.x, t0.y * k0.y);
    a.u[1] = pk(t0.z * k0.z, t0.w * k0.w);
    a.u[2] = pk(t1.x * k1.x, t1.y * k1.y);
    a.u[3] = pk(t1.z * k1.z, t1.w * k1.w);
    a.u[4] = 0; a.u[5] = 0; a.u[6] = 0; a.u[7] = 0;

    v8f cv = {};
    cv = wmma_bf16(a.v, bHA, cv);
    *(uint4*)(vt + m * 32 + (hf << 3)) = pack8(cv);   // V^T, left half

    v8f co = {};
    co = wmma_bf16(aHA, load_Bt(vt, 0, 0, 32), co);

    const float bf = bias2[f];
    u16* ab = act + (size_t)b * 4096 + f * 64;
#pragma unroll
    for (int i = 0; i < 4; ++i) {
      float r0v = fmaxf(co[2 * i] + bf, 0.f);
      float r1v = fmaxf(co[2 * i + 1] + bf, 0.f);
      float mv = max_nb(fmaxf(r0v, r1v));
      if ((m & 1) == 0) {
        const int pi = (hf << 2) + i, pj = m >> 1;
        ab[pi * 8 + pj] = f2bf(mv);
      }
    }
  }
}

// ---------------------------------------------------------------------------
// FC1: h = relu(act(2048x4096) * fc1_w^T + b). Block tile 64x32, 8 waves,
// K=32 steps; A row-major LDS, B staged N-major (fc1_w native layout).
// ---------------------------------------------------------------------------
__global__ __launch_bounds__(256) void fc1_kernel(
    const u16* __restrict__ act, const u16* __restrict__ Wb,
    const float* __restrict__ fc1_b, u16* __restrict__ h) {
  __shared__ __align__(16) u16 Ab[64 * 32];
  __shared__ __align__(16) u16 Bt[32 * 32];
  const int t = threadIdx.x, wave = t >> 5, lane = t & 31;
  const int mb = blockIdx.y * 64, nb = blockIdx.x * 32;
  const int tm = (wave >> 1) << 4;   // 0,16,32,48
  const int tn = (wave & 1) << 4;    // 0,16
  const int ar = t >> 2, ac = (t & 3) << 3;

  v8f c = {};
  for (int kk = 0; kk < 4096; kk += 32) {
    *(uint4*)(Ab + ar * 32 + ac) =
        *(const uint4*)(act + (size_t)(mb + ar) * 4096 + kk + ac);
    if (t < 128)
      *(uint4*)(Bt + (t >> 2) * 32 + ac) =
          *(const uint4*)(Wb + (size_t)(nb + (t >> 2)) * 4096 + kk + ac);
    __syncthreads();
    c = wmma_bf16(load_A(Ab, tm, 0, 32), load_Bt(Bt, tn, 0, 32), c);
    __syncthreads();
  }
  const int cn = nb + tn + (lane & 15);
  const int crow = mb + tm + ((lane >> 4) << 3);
  const float bf = fc1_b[cn];
#pragma unroll
  for (int v = 0; v < 8; ++v) {
    float o = c[v] + bf;
    h[(size_t)(crow + v) * 512 + cn] = f2bf(o > 0.f ? o : 0.f);
  }
}

// ---------------------------------------------------------------------------
// FC2 + softmax: per wave one 16-row tile, K=512 (16 WMMA), N padded 10->16.
// fc2_w native (10,512) layout is already N-major/B-friendly.
// ---------------------------------------------------------------------------
__global__ __launch_bounds__(128) void fc2_softmax_kernel(
    const u16* __restrict__ h, const float* __restrict__ fc2_w,
    const float* __restrict__ fc2_b, float* __restrict__ out) {
  __shared__ __align__(16) float L[4][16 * 16];
  const int t = threadIdx.x, wave = t >> 5, lane = t & 31;
  const int r0 = (blockIdx.x * 4 + wave) * 16;
  const int m = lane & 15, hf = lane >> 4;

  v8f c = {};
  for (int kk = 0; kk < 512; kk += 32) {
    BfVec a, b;
#pragma unroll
    for (int v = 0; v < 8; ++v) {
      const int kb = kk + ((v & 4) << 2) + hf * 8 + ((v & 3) << 1);
      a.u[v] = *(const unsigned*)(h + (size_t)(r0 + m) * 512 + kb);
      const int kbb = kk + ((v & 4) << 2) + hf * 8 + ((v & 3) << 1);
      if (m < 10) {
        const float2 wv = *(const float2*)(fc2_w + m * 512 + kbb);
        b.u[v] = pk(wv.x, wv.y);
      } else {
        b.u[v] = 0;
      }
    }
    c = wmma_bf16(a.v, b.v, c);
  }
  float* Lw = L[wave];
  const float bf = (m < 10) ? fc2_b[m] : 0.f;
#pragma unroll
  for (int v = 0; v < 8; ++v) Lw[(hf * 8 + v) * 16 + m] = c[v] + bf;
  __syncthreads();
  if (lane < 16) {
    float mx = -1e30f;
    for (int j = 0; j < 10; ++j) mx = fmaxf(mx, Lw[lane * 16 + j]);
    float e[10], s = 0.f;
    for (int j = 0; j < 10; ++j) { e[j] = __expf(Lw[lane * 16 + j] - mx); s += e[j]; }
    const float inv = 1.f / s;
    float* op = out + (size_t)(r0 + lane) * 10;
    for (int j = 0; j < 10; ++j) op[j] = e[j] * inv;
  }
}

// ---------------------------------------------------------------------------
extern "C" void kernel_launch(void* const* d_in, const int* in_sizes, int n_in,
                              void* d_out, int out_size, void* d_ws, size_t ws_size,
                              hipStream_t stream) {
  const float* x      = (const float*)d_in[0];   // (2048,3,32,32)
  const float* f1     = (const float*)d_in[1];   // (32,3,3)
  const float* b1     = (const float*)d_in[2];   // (32,1)
  const float* f2     = (const float*)d_in[3];   // (64,3,3)
  const float* b2     = (const float*)d_in[4];   // (64,1)
  const float* fc1_w  = (const float*)d_in[5];   // (512,4096)
  const float* fc1_b  = (const float*)d_in[6];   // (512,)
  const float* fc2_w  = (const float*)d_in[7];   // (10,512)
  const float* fc2_b  = (const float*)d_in[8];   // (10,)
  float* out = (float*)d_out;                    // (2048,10) f32

  char* ws = (char*)d_ws;
  float* K1c   = (float*)(ws + 0);                 //  128 KB
  float* K2c   = (float*)(ws + 131072);            //   64 KB
  float* psum1 = (float*)(ws + 196608);            //    2 MB
  u16*   act   = (u16*)(ws + 2293760);             //   16 MB bf16
  u16*   Wb    = (u16*)(ws + 19070976);            //    4 MB bf16
  u16*   hbuf  = (u16*)(ws + 23265280);            //    2 MB bf16

  prep_filters_kernel<<<192, 256, 0, stream>>>(f1, f2, K1c, K2c);
  prep_weights_kernel<<<1024, 256, 0, stream>>>(fc1_w, Wb);
  stage1_kernel<<<2048, 128, 0, stream>>>(x, K1c, b1, psum1);
  stage2_kernel<<<2048, 128, 0, stream>>>(psum1, K2c, b2, act);
  fc1_kernel<<<dim3(16, 32), 256, 0, stream>>>(act, Wb, fc1_b, hbuf);
  fc2_softmax_kernel<<<32, 128, 0, stream>>>(hbuf, fc2_w, fc2_b, out);
}
